// SSMLayer_549755814622
// MI455X (gfx1250) — compile-verified, matrix operand order checked by reading, not run on hardware
//
#include <hip/hip_runtime.h>
#include <math.h>

typedef __attribute__((ext_vector_type(16))) _Float16     v16h;
typedef __attribute__((ext_vector_type(8)))  float        v8f;
typedef __attribute__((ext_vector_type(4)))  unsigned int u32x4;
typedef __attribute__((ext_vector_type(8)))  unsigned int u32x8;

#define BATCH 1024
#define LSEQ  256
#define SDIM  256
#define DIN   512
#define KFLAT (LSEQ * DIN)        // 131072
#define NROWS (BATCH * LSEQ)      // 262144
#define EPS   1e-6f

// d_out layout (floats): y_hat [NROWS] | alpha [BATCH*257*256] | Sigma [NROWS]
#define YHAT_OFF   0
#define ALPHA_OFF  ((size_t)NROWS)
#define ALPHA_CNT  ((size_t)BATCH * 257 * 256)
#define SIGMA_OFF  (ALPHA_OFF + ALPHA_CNT)

// ws layout (floats)
#define WS_MEAN1   0          // 262144
#define WS_RSTD1   262144     // 262144
#define WS_MEAN2   524288     // 1024
#define WS_RSTD2   525312     // 1024
#define WS_GWC     526336     // 256
#define WS_BWC     526592     // 256
#define WS_GW2     526848     // 256  (atomic accum)
#define WS_BW2     527104     // 256  (atomic accum)
#define WS_W1S     527360     // 2
#define WS_A0RAW   527488     // 262144 (atomic accum)
#define WS_ZERO_BEG 526848
#define WS_ZERO_CNT (527488 + 262144 - 526848)   // 262784

// ---------------------------------------------------------------------------
// TDM: async 2D tile DMA global->LDS (CDNA5 ISA).
// D# group0: [count=1 | lds_addr | global_addr_lo | addr_hi[24:0] + type=2]
// D# group1: wg_mask=0, data_size=4B, tensor_dim0=row_stride, tensor_dim1=2^20
//            (tiles always in-bounds), tile_dim0=w, tile_dim1=h, dim0_stride.
// Issued by one wave; completion via s_wait_tensorcnt + workgroup barrier.
// ---------------------------------------------------------------------------
__device__ __forceinline__ void tdm_load_2d_f32(unsigned lds_off, const float* gaddr,
                                                unsigned tile_w, unsigned tile_h,
                                                unsigned row_stride) {
    unsigned long long ga = (unsigned long long)(uintptr_t)gaddr;
    u32x4 d0;
    d0[0] = 1u;                                                  // count=1 (valid user D#)
    d0[1] = lds_off;                                             // lds_addr (bytes)
    d0[2] = (unsigned)ga;                                        // global_addr[31:0]
    d0[3] = ((unsigned)(ga >> 32) & 0x01FFFFFFu) | (2u << 30);   // addr[56:32] | type=2
    u32x8 d1;
    d1[0] = 2u << 16;                                            // data_size=2 (4 bytes)
    d1[1] = (row_stride & 0xFFFFu) << 16;                        // tensor_dim0 lo16
    d1[2] = (row_stride >> 16) & 0xFFFFu;                        // tensor_dim0 hi | dim1 lo=0
    d1[3] = 0x10u | (tile_w << 16);                              // dim1 hi (2^20) | tile_dim0
    d1[4] = tile_h & 0xFFFFu;                                    // tile_dim1 (tile_dim2=0)
    d1[5] = row_stride;                                          // tensor_dim0_stride lo32
    d1[6] = 0u;                                                  // stride hi / dim1_stride lo
    d1[7] = 0u;
    asm volatile("tensor_load_to_lds %0, %1" :: "s"(d0), "s"(d1) : "memory");
}

// ---------------------------------------------------------------------------
// Fragment gather for 16-bit A/B matrices (ISA 7.12.2 layout):
// lane: M/N = lane%16, hi = lane/16. Element pair p:
//   p<4 : K = hi*8 + 2p + {0,1}       p>=4: K = 16 + hi*8 + 2(p-4) + {0,1}
// ---------------------------------------------------------------------------
__device__ __forceinline__ v16h load_frag(const _Float16* base, int hi) {
    v16h f;
#pragma unroll
    for (int p = 0; p < 8; ++p) {
        int k0 = ((p < 4) ? (2 * p) : (16 + 2 * (p - 4))) + hi * 8;
        f[2 * p]     = base[k0];
        f[2 * p + 1] = base[k0 + 1];
    }
    return f;
}

__device__ __forceinline__ v16h load_frag_f32(const float* base, int hi) {
    v16h f;
#pragma unroll
    for (int p = 0; p < 8; ++p) {
        int k0 = ((p < 4) ? (2 * p) : (16 + 2 * (p - 4))) + hi * 8;
        f[2 * p]     = (_Float16)base[k0];
        f[2 * p + 1] = (_Float16)base[k0 + 1];
    }
    return f;
}

__device__ __forceinline__ float hsig_c(float pre) {
    // -0.5 + hardsigmoid(pre) == clamp(pre/6, -0.5, 0.5)
    return fminf(fmaxf(pre * (1.0f / 6.0f), -0.5f), 0.5f);
}

// ---------------------------------------------------------------------------
__global__ void k_zero(float* p, int n) {
    int i = blockIdx.x * blockDim.x + threadIdx.x;
    if (i < n) p[i] = 0.0f;
}

// Gw1 = sum g1*w1 ; Bw1 = sum b1*w1 + w1_b
__global__ __launch_bounds__(256) void k_w1sums(const float* g1, const float* b1,
                                                const float* w1W, const float* w1b,
                                                float* W1S) {
    int tid = threadIdx.x;
    float gw = 0.f, bw = 0.f;
    for (int i = tid; i < DIN; i += 256) {
        float w = w1W[i];
        gw += g1[i] * w;
        bw += b1[i] * w;
    }
    for (int off = 16; off; off >>= 1) {
        gw += __shfl_xor(gw, off, 32);
        bw += __shfl_xor(bw, off, 32);
    }
    __shared__ float pg[8], pb[8];
    int lane = tid & 31, wv = tid >> 5;
    if (lane == 0) { pg[wv] = gw; pb[wv] = bw; }
    __syncthreads();
    if (tid == 0) {
        float G = 0.f, Bv = 0.f;
        for (int i = 0; i < 8; ++i) { G += pg[i]; Bv += pb[i]; }
        W1S[0] = G;
        W1S[1] = Bv + w1b[0];
    }
}

// Column sums for the c-GEMM weight (K=512, N=256)
__global__ __launch_bounds__(256) void k_colsums_c(const float* Wc, const float* gc,
                                                   const float* bc, const float* wcb,
                                                   float* GwC, float* BwC) {
    int j = threadIdx.x;
    float gw = 0.f, bw = 0.f;
    for (int k = 0; k < DIN; ++k) {
        float w = Wc[(size_t)k * SDIM + j];
        gw += gc[k] * w;
        bw += bc[k] * w;
    }
    GwC[j] = gw;
    BwC[j] = bw + wcb[j];
}

// Column sums for the a0-GEMM weight (K=131072, N=256): 512 K-chunk blocks
__global__ __launch_bounds__(256) void k_colsums_a(const float* Wa, const float* g2,
                                                   const float* b2, float* Gw2, float* Bw2) {
    int j  = threadIdx.x;
    int kb = blockIdx.x * 256;
    float gw = 0.f, bw = 0.f;
    for (int kk = 0; kk < 256; ++kk) {
        int k = kb + kk;
        float w = Wa[(size_t)k * SDIM + j];
        gw += g2[k] * w;
        bw += b2[k] * w;
    }
    unsafeAtomicAdd(&Gw2[j], gw);
    unsafeAtomicAdd(&Bw2[j], bw);
}

// Per-row (512-wide) mean/rstd + fused Sigma = softplus(LN(x)@w1 + w1_b).
__global__ __launch_bounds__(256) void k_rowstats(const float* x, const float* g1,
                                                  const float* w1W, const float* W1S,
                                                  float* mean1, float* rstd1, float* sigma) {
    __shared__ float gw[DIN];
    int tid = threadIdx.x;
    for (int i = tid; i < DIN; i += 256) gw[i] = g1[i] * w1W[i];
    __syncthreads();
    int lane = tid & 31, wv = tid >> 5;
    int row  = blockIdx.x * 8 + wv;
    const float* xr = x + (size_t)row * DIN;
    float s = 0.f, ss = 0.f, dw = 0.f;
    for (int i = lane; i < DIN; i += 32) {
        float v = xr[i];
        s += v; ss += v * v; dw += v * gw[i];
    }
    for (int off = 16; off; off >>= 1) {
        s  += __shfl_xor(s,  off, 32);
        ss += __shfl_xor(ss, off, 32);
        dw += __shfl_xor(dw, off, 32);
    }
    if (lane == 0) {
        float m   = s * (1.0f / DIN);
        float var = ss * (1.0f / DIN) - m * m;
        float rs  = rsqrtf(var + EPS);
        mean1[row] = m;
        rstd1[row] = rs;
        float z = (dw - m * W1S[0]) * rs + W1S[1];
        sigma[row] = (z > 30.f) ? z : log1pf(__expf(z));
    }
}

// Per-batch (131072-wide) mean/rstd
__global__ __launch_bounds__(256) void k_batchstats(const float* x, float* mean2, float* rstd2) {
    int b = blockIdx.x, tid = threadIdx.x;
    const float* xb = x + (size_t)b * KFLAT;
    float s = 0.f, ss = 0.f;
    for (int i = tid; i < KFLAT; i += 256) {
        float v = xb[i];
        s += v; ss += v * v;
    }
    for (int off = 16; off; off >>= 1) {
        s  += __shfl_xor(s,  off, 32);
        ss += __shfl_xor(ss, off, 32);
    }
    __shared__ float ps[8], pss[8];
    int lane = tid & 31, wv = tid >> 5;
    if (lane == 0) { ps[wv] = s; pss[wv] = ss; }
    __syncthreads();
    if (tid == 0) {
        float S = 0.f, SS = 0.f;
        for (int i = 0; i < 8; ++i) { S += ps[i]; SS += pss[i]; }
        float m   = S * (1.0f / KFLAT);
        float var = SS * (1.0f / KFLAT) - m * m;
        mean2[b] = m;
        rstd2[b] = rsqrtf(var + EPS);
    }
}

// ---------------------------------------------------------------------------
// c-GEMM: [262144 x 512] @ [512 x 256], f16 WMMA, LN folded in epilogue.
// Block tile 32 x 256; 8 waves, wave w: M-half = w&1, N-quad = w>>1 (4 tiles).
// A tile (32x32 f32) DMA'd by the Tensor Data Mover straight into LDS;
// converted f32->f16 at fragment build. B tile staged f16 with g-scale.
// Result c[b,l,:] staged at d_out alpha slot [b, l+1, :].
// ---------------------------------------------------------------------------
__global__ __launch_bounds__(256) void k_cgemm(const float* x, const float* Wc, const float* gc,
                                               const float* mean1, const float* rstd1,
                                               const float* GwC, const float* BwC,
                                               float* alphaBase) {
    __shared__ float    Af[32][32];     // [row][k]  (TDM destination)
    __shared__ _Float16 Bh[SDIM][32];   // [col][k]
    const int tid  = threadIdx.x;
    const int lane = tid & 31, wv = tid >> 5;
    const int row0 = blockIdx.x * 32;
    const int m    = lane & 15, hi = lane >> 4;
    const int mh   = wv & 1;            // M half (16 rows)
    const int n0   = (wv >> 1) * 64;    // N quad base
    const unsigned ldsA = (unsigned)(uintptr_t)&Af[0][0];

    v8f acc[4];
#pragma unroll
    for (int t = 0; t < 4; ++t) acc[t] = (v8f){0, 0, 0, 0, 0, 0, 0, 0};

    for (int kc = 0; kc < DIN; kc += 32) {
        __syncthreads();   // previous fragments consumed; LDS safe to overwrite
        if (tid < 32) {    // wave 0 drives the TDM
            tdm_load_2d_f32(ldsA, x + (size_t)row0 * DIN + kc, 32, 32, DIN);
        }
        // stage B tile: col = tid (coalesced), 32 K-halfs each, scaled by gc
#pragma unroll 8
        for (int kk = 0; kk < 32; ++kk) {
            float w = Wc[(size_t)(kc + kk) * SDIM + tid] * gc[kc + kk];
            Bh[tid][kk] = (_Float16)w;
        }
        __builtin_prefetch(&Wc[(size_t)(kc + 32) * SDIM + tid], 0, 1);  // next K-slab
        if (tid < 32) __builtin_amdgcn_s_wait_tensorcnt(0);
        __syncthreads();

        v16h af = load_frag_f32(&Af[mh * 16 + m][0], hi);
#pragma unroll
        for (int t = 0; t < 4; ++t) {
            v16h bf = load_frag(&Bh[n0 + t * 16 + m][0], hi);
            acc[t] = __builtin_amdgcn_wmma_f32_16x16x32_f16(false, af, false, bf,
                                                            (short)0, acc[t], false, false);
        }
    }

#pragma unroll
    for (int t = 0; t < 4; ++t) {
        int col = n0 + t * 16 + m;
        float gwc = GwC[col], bwc = BwC[col];
#pragma unroll
        for (int e = 0; e < 8; ++e) {
            int row = row0 + mh * 16 + e + hi * 8;
            float mn = mean1[row], rs = rstd1[row];
            int b = row >> 8;                              // row = b*256 + l
            size_t slot = (size_t)(row + b + 1) * 256;     // alpha[b, l+1, :]
            alphaBase[slot + col] = hsig_c((acc[t][e] - mn * gwc) * rs + bwc);
        }
    }
}

// a0-GEMM: [1024 x 131072] @ [131072 x 256], split-K (64 chunks of 2048),
// f16 WMMA, f32 atomic accumulation into a0raw.
__global__ __launch_bounds__(256) void k_agemm(const float* x, const float* Wa, const float* g2,
                                               float* a0raw) {
    __shared__ _Float16 Ah[16][32];
    __shared__ _Float16 Bh[SDIM][32];
    const int tid   = threadIdx.x;
    const int lane  = tid & 31, wv = tid >> 5;
    const int row0  = blockIdx.x * 16;
    const int kbase = blockIdx.y * 2048;
    const int m     = lane & 15, hi = lane >> 4;
    const int n0    = wv * 32;
    v8f acc0 = {0, 0, 0, 0, 0, 0, 0, 0};
    v8f acc1 = {0, 0, 0, 0, 0, 0, 0, 0};

    for (int kc = 0; kc < 2048; kc += 32) {
        int k0 = kbase + kc;
        {
            int idx = tid * 2;
            int r = idx >> 5, kk = idx & 31;
            const float* xp = x + (size_t)(row0 + r) * KFLAT + k0 + kk;
            Ah[r][kk]     = (_Float16)xp[0];
            Ah[r][kk + 1] = (_Float16)xp[1];
        }
#pragma unroll 8
        for (int kk = 0; kk < 32; ++kk) {
            float w = Wa[(size_t)(k0 + kk) * SDIM + tid] * g2[k0 + kk];
            Bh[tid][kk] = (_Float16)w;
        }
        __builtin_prefetch(&Wa[(size_t)(k0 + 32) * SDIM + tid], 0, 1);
        __syncthreads();
        v16h af  = load_frag(&Ah[m][0], hi);
        v16h bf0 = load_frag(&Bh[n0 + m][0], hi);
        v16h bf1 = load_frag(&Bh[n0 + 16 + m][0], hi);
        acc0 = __builtin_amdgcn_wmma_f32_16x16x32_f16(false, af, false, bf0, (short)0, acc0, false, false);
        acc1 = __builtin_amdgcn_wmma_f32_16x16x32_f16(false, af, false, bf1, (short)0, acc1, false, false);
        __syncthreads();
    }

    int col0 = n0 + m, col1 = n0 + 16 + m;
#pragma unroll
    for (int e = 0; e < 8; ++e) {
        int row = row0 + e + hi * 8;
        unsafeAtomicAdd(&a0raw[row * 256 + col0], acc0[e]);
        unsafeAtomicAdd(&a0raw[row * 256 + col1], acc1[e]);
    }
}

// Finalize a0: LN-fold + hardsigmoid; write straight into alpha[b,0,:]
__global__ __launch_bounds__(256) void k_a0fin(const float* a0raw, const float* mean2,
                                               const float* rstd2, const float* Gw2,
                                               const float* Bw2, const float* wab,
                                               float* alphaBase) {
    int b = blockIdx.x, j = threadIdx.x;
    float raw = a0raw[b * 256 + j];
    float pre = (raw - mean2[b] * Gw2[j]) * rstd2[b] + Bw2[j] + wab[j];
    alphaBase[(size_t)b * 257 * 256 + j] = hsig_c(pre);
}

// Sequential recurrence: a'[0]=a[0]+c[0]; a'[1]=-sum_{k>=1}a[k]+c[1]; a'[j]=a[j-1]+c[j].
// One block per batch; thread j owns state j; c read in place and overwritten by alpha.
__global__ __launch_bounds__(256) void k_recur(float* out) {
    const int b    = blockIdx.x;
    const int j    = threadIdx.x;
    const int lane = j & 31, wv = j >> 5;
    __shared__ float a_lds[256];
    __shared__ float wsum[8];
    float* alpha = out + ALPHA_OFF + (size_t)b * 257 * 256;
    float* yhat  = out + YHAT_OFF + (size_t)b * LSEQ;

    float a = alpha[j];     // a0 written by k_a0fin
    a_lds[j] = a;
    __syncthreads();

    for (int l = 0; l < LSEQ; ++l) {
        float cj    = alpha[(size_t)(l + 1) * 256 + j];     // staged c[b,l,j]
        float aprev = (j >= 2) ? a_lds[j - 1] : 0.0f;
        float msk   = (j >= 1) ? a : 0.0f;
        for (int off = 16; off; off >>= 1) msk += __shfl_xor(msk, off, 32);
        if (lane == 0) wsum[wv] = msk;
        __syncthreads();                                    // syncA
        float s = 0.f;
#pragma unroll
        for (int i = 0; i < 8; ++i) s += wsum[i];
        float na;
        if (j == 0)      na = a + cj;
        else if (j == 1) na = -s + cj;
        else             na = aprev + cj;
        a_lds[j] = na;
        a = na;
        alpha[(size_t)(l + 1) * 256 + j] = na;              // overwrite staged c
        __syncthreads();                                    // syncB
        if (j == 0) yhat[l] = a_lds[0] + a_lds[1];
    }
}

// ---------------------------------------------------------------------------
extern "C" void kernel_launch(void* const* d_in, const int* in_sizes, int n_in,
                              void* d_out, int out_size, void* d_ws, size_t ws_size,
                              hipStream_t stream) {
    const float* x   = (const float*)d_in[0];
    const float* w1W = (const float*)d_in[1];
    const float* w1b = (const float*)d_in[2];
    const float* g1  = (const float*)d_in[3];
    const float* b1  = (const float*)d_in[4];
    const float* WaW = (const float*)d_in[5];
    const float* wab = (const float*)d_in[6];
    const float* g2  = (const float*)d_in[7];
    const float* b2  = (const float*)d_in[8];
    const float* WcW = (const float*)d_in[9];
    const float* wcb = (const float*)d_in[10];
    const float* gc  = (const float*)d_in[11];
    const float* bc  = (const float*)d_in[12];

    float* out = (float*)d_out;
    float* ws  = (float*)d_ws;

    float* mean1 = ws + WS_MEAN1;
    float* rstd1 = ws + WS_RSTD1;
    float* mean2 = ws + WS_MEAN2;
    float* rstd2 = ws + WS_RSTD2;
    float* GwC   = ws + WS_GWC;
    float* BwC   = ws + WS_BWC;
    float* Gw2   = ws + WS_GW2;
    float* Bw2   = ws + WS_BW2;
    float* W1S   = ws + WS_W1S;
    float* a0raw = ws + WS_A0RAW;

    float* alphaBase = out + ALPHA_OFF;
    float* sigma     = out + SIGMA_OFF;

    // 1. zero the atomic accumulators (Gw2/Bw2/a0raw)
    k_zero<<<dim3((WS_ZERO_CNT + 255) / 256), dim3(256), 0, stream>>>(ws + WS_ZERO_BEG, WS_ZERO_CNT);
    // 2. scalar weight sums for Sigma path
    k_w1sums<<<dim3(1), dim3(256), 0, stream>>>(g1, b1, w1W, w1b, W1S);
    // 3. column sums for c-GEMM LN fold
    k_colsums_c<<<dim3(1), dim3(256), 0, stream>>>(WcW, gc, bc, wcb, GwC, BwC);
    // 4. column sums for a0-GEMM LN fold (atomic)
    k_colsums_a<<<dim3(KFLAT / 256), dim3(256), 0, stream>>>(WaW, g2, b2, Gw2, Bw2);
    // 5. per-row stats + Sigma
    k_rowstats<<<dim3(NROWS / 8), dim3(256), 0, stream>>>(x, g1, w1W, W1S, mean1, rstd1, sigma);
    // 6. per-batch stats
    k_batchstats<<<dim3(BATCH), dim3(256), 0, stream>>>(x, mean2, rstd2);
    // 7. c GEMM (WMMA + TDM A-tiles) -> staged into alpha[:,1:,:]
    k_cgemm<<<dim3(NROWS / 32), dim3(256), 0, stream>>>(x, WcW, gc, mean1, rstd1, GwC, BwC, alphaBase);
    // 8. a0 GEMM (WMMA, split-K, atomic accumulate)
    k_agemm<<<dim3(BATCH / 16, KFLAT / 2048), dim3(256), 0, stream>>>(x, WaW, g2, a0raw);
    // 9. finalize a0 -> alpha[:,0,:]
    k_a0fin<<<dim3(BATCH), dim3(256), 0, stream>>>(a0raw, mean2, rstd2, Gw2, Bw2, wab, alphaBase);
    // 10. sequential recurrence -> alpha[:,1:,:] (in place over staged c) + y_hat
    k_recur<<<dim3(BATCH), dim3(256), 0, stream>>>(out);
}
